// Spiking_QK_WindowAttention3D_44650480009378
// MI455X (gfx1250) — matrix operand order repaired and checked by reading, not compiled, not run on hardware
//
#include <hip/hip_runtime.h>
#include <hip/hip_bf16.h>

typedef __attribute__((ext_vector_type(16))) _Float16 v16h;
typedef __attribute__((ext_vector_type(8)))  _Float16 v8h;
typedef __attribute__((ext_vector_type(8)))  float    v8f;

// dims
#define T_   4
#define B_   1024
#define HW_  64
#define C_   256
#define RPT  (B_ * HW_)          // rows per timestep = 65536
#define NROW 16                  // rows per block tile

// scratch layout in halves
#define WQ_OFF   0
#define WK_OFF   65536
#define WP_OFF   131072
#define POS_OFF  196608
#define ATTN_OFF 262144          // 67108864 halves follow

__device__ __forceinline__ float lif_step(float& v, float pre) {
    v += (pre - v) * 0.5f;                 // tau = 2
    float s = (v >= 1.0f) ? 1.0f : 0.0f;   // v_th = 1
    v *= (1.0f - s);                       // hard reset
    return s;
}

// ---------------- kernel 1: f32 -> f16 weight/pos conversion -----------------
__global__ __launch_bounds__(256) void cvt_kernel(
    const float* __restrict__ Wq, const float* __restrict__ Wk,
    const float* __restrict__ Wp, const float* __restrict__ pos,
    _Float16* __restrict__ ws)
{
    int g = blockIdx.x * 256 + threadIdx.x;      // 0 .. 262143
    int which = g >> 16, off = g & 65535;
    float v;
    if      (which == 0) v = Wq[off];
    else if (which == 1) v = Wk[off];
    else if (which == 2) v = Wp[off];
    else                 v = pos[off];
    ws[g] = (_Float16)v;
}

// --------- kernel 2: input-LIF + Q/K WMMA GEMMs + att LIF + attn ------------
__global__ __launch_bounds__(256) void qk_kernel(
    const float* __restrict__ x,
    const _Float16* __restrict__ Wq, const _Float16* __restrict__ Wk,
    const _Float16* __restrict__ pos, _Float16* __restrict__ attn)
{
    __shared__ alignas(32) _Float16 xsT[T_][NROW][C_];   // 32 KB spike tile
    __shared__ float attsum[NROW][8];
    __shared__ float attv  [NROW][8];
    __shared__ _Float16 atts[NROW][8];

    const int tid  = threadIdx.x;
    const int lane = tid & 31;
    const int wave = tid >> 5;                 // 0..3: Q, 4..7: K
    const int rbase = blockIdx.x * NROW;

    // ---- phase 0: input LIF (each thread owns 16 consecutive channels) ----
    {
        int e = tid * 16, row = e >> 8, c0 = e & 255;
        float v[16];
#pragma unroll
        for (int k = 0; k < 16; ++k) v[k] = 0.f;
#pragma unroll
        for (int t = 0; t < T_; ++t) {
            const float4* xp = (const float4*)(x + ((size_t)(t * RPT + rbase + row)) * C_ + c0);
#pragma unroll
            for (int q4 = 0; q4 < 4; ++q4) {
                float4 xv = xp[q4];
                float xa[4] = {xv.x, xv.y, xv.z, xv.w};
#pragma unroll
                for (int j = 0; j < 4; ++j) {
                    int k = q4 * 4 + j;
                    xsT[t][row][c0 + k] = (_Float16)lif_step(v[k], xa[j]);
                }
            }
        }
        if (tid < 128) attv[tid >> 3][tid & 7] = 0.f;
    }

    const _Float16* Wsel = (wave < 4) ? Wq : Wk;
    const int nb = (wave & 3) * 64;            // 4 tiles of 16 columns

    v8f vstate[4];                             // persistent membrane (q or k)
#pragma unroll
    for (int i = 0; i < 4; ++i) vstate[i] = (v8f)0.f;
    v8f sk[4];                                 // k spikes of current t

    for (int t = 0; t < T_; ++t) {
        __syncthreads();
        if (tid < 128) attsum[tid >> 3][tid & 7] = 0.f;
        __syncthreads();

        // ---- GEMM: 16x256 spikes x 256x64 weights, K in 8 steps of 32 ----
        v8f acc[4];
#pragma unroll
        for (int i = 0; i < 4; ++i) acc[i] = (v8f)0.f;

        for (int kb = 0; kb < 8; ++kb) {
            const int k0 = kb * 32;
            v16h a;                            // A frag: ISA 16-bit 16x32 layout
            {
                int arow = lane & 15;
                int koff = k0 + ((lane < 16) ? 0 : 8);
                v8h a0 = *(const v8h*)&xsT[t][arow][koff];
                v8h a1 = *(const v8h*)&xsT[t][arow][koff + 16];
#pragma unroll
                for (int i = 0; i < 8; ++i) { a[i] = a0[i]; a[i + 8] = a1[i]; }
            }
#pragma unroll
            for (int i = 0; i < 4; ++i) {
                int n = nb + i * 16 + (lane & 15);
                const _Float16* bp_ = &Wsel[n * C_ + k0 + ((lane < 16) ? 0 : 8)];
                v8h b0 = *(const v8h*)bp_;
                v8h b1 = *(const v8h*)(bp_ + 16);
                v16h bfr;
#pragma unroll
                for (int q = 0; q < 8; ++q) { bfr[q] = b0[q]; bfr[q + 8] = b1[q]; }
                acc[i] = __builtin_amdgcn_wmma_f32_16x16x32_f16(
                    false, a, false, bfr, (short)0, acc[i], false, false);
            }
        }

        if (wave < 4) {
            // ---- Q path: LIF, then 32-channel group sums via LDS atomics ----
#pragma unroll
            for (int i = 0; i < 4; ++i) {
                int g = (nb + i * 16) >> 5;
#pragma unroll
                for (int j = 0; j < 8; ++j) {
                    int M = j + ((lane < 16) ? 0 : 8);
                    float vv = vstate[i][j];
                    float s = lif_step(vv, acc[i][j]);
                    vstate[i][j] = vv;
                    atomicAdd(&attsum[M][g], s);
                }
            }
        } else {
            // ---- K path: add positional encoding, LIF ----
#pragma unroll
            for (int i = 0; i < 4; ++i) {
                int N = nb + i * 16 + (lane & 15);
#pragma unroll
                for (int j = 0; j < 8; ++j) {
                    int M  = j + ((lane < 16) ? 0 : 8);
                    int hw = (rbase + M) & 63;
                    float pre = acc[i][j] + (float)pos[t * 16384 + hw * C_ + N];
                    float vv = vstate[i][j];
                    float s = lif_step(vv, pre);
                    vstate[i][j] = vv;
                    sk[i][j] = s;
                }
            }
        }
        __syncthreads();

        // ---- token-attention LIF on group sums ----
        if (tid < 128) {
            int rr = tid >> 3, g = tid & 7;
            float vv = attv[rr][g];
            float s = lif_step(vv, attsum[rr][g]);
            attv[rr][g] = vv;
            atts[rr][g] = (_Float16)s;
        }
        __syncthreads();

        // ---- attn = k_spike * att_spike, stored f16 flat (t,r,c) ----
        if (wave >= 4) {
#pragma unroll
            for (int i = 0; i < 4; ++i) {
                int n0 = nb + i * 16;
                int g  = n0 >> 5;
                int N  = n0 + (lane & 15);
#pragma unroll
                for (int j = 0; j < 8; ++j) {
                    int M = j + ((lane < 16) ? 0 : 8);
                    float av = (float)atts[M][g];
                    attn[((size_t)(t * RPT + rbase + M)) * C_ + N] =
                        (_Float16)(sk[i][j] * av);
                }
            }
        }
    }
}

// --------- kernel 3: gather x2, attn_out LIF, proj WMMA GEMM + bias ---------
__global__ __launch_bounds__(256) void proj_kernel(
    const _Float16* __restrict__ attn, const _Float16* __restrict__ Wp,
    const float* __restrict__ bp,
    float* __restrict__ y, float* __restrict__ attn_out)
{
    __shared__ alignas(32) _Float16 x2T[NROW][C_];

    const int tid = threadIdx.x;
    const int lane = tid & 31;
    const int wave = tid >> 5;
    const int rbase = blockIdx.x * NROW;

    // each thread owns 16 consecutive channels of one row (same nh group)
    const int e = tid * 16, row = e >> 8, c0 = e & 255;
    const int r = rbase + row;
    const int b = r >> 6, hw = r & 63, h = hw >> 3, w = hw & 7;
    const int nh = c0 >> 5;
    const size_t srcbase = (size_t)b * 65536 + (size_t)nh * 8192 +
                           h * 256 + w * 32 + (c0 & 31);

    float v[16];
#pragma unroll
    for (int k = 0; k < 16; ++k) v[k] = 0.f;

    for (int t = 0; t < T_; ++t) {
        __syncthreads();
        // gather 32B-aligned 16-half segments (the permute collapses to this)
        {
            const float4* sp = (const float4*)(attn + srcbase + t * 2048);
            float4* dp = (float4*)&x2T[row][c0];
            dp[0] = sp[0];
            dp[1] = sp[1];
        }
        __syncthreads();

        // attn_out = lif(x2)
#pragma unroll
        for (int k = 0; k < 16; ++k) {
            float xv = (float)x2T[row][c0 + k];
            attn_out[((size_t)(t * RPT + r)) * C_ + c0 + k] = lif_step(v[k], xv);
        }

        // y = x2 @ Wp^T + bp   (wave handles 2 column tiles of 16)
        v8f acc[2];
        acc[0] = (v8f)0.f; acc[1] = (v8f)0.f;
        for (int kb = 0; kb < 8; ++kb) {
            const int k0 = kb * 32;
            v16h a;
            {
                int arow = lane & 15;
                int koff = k0 + ((lane < 16) ? 0 : 8);
                v8h a0 = *(const v8h*)&x2T[arow][koff];
                v8h a1 = *(const v8h*)&x2T[arow][koff + 16];
#pragma unroll
                for (int i = 0; i < 8; ++i) { a[i] = a0[i]; a[i + 8] = a1[i]; }
            }
#pragma unroll
            for (int i = 0; i < 2; ++i) {
                int n = wave * 32 + i * 16 + (lane & 15);
                const _Float16* bp_ = &Wp[n * C_ + k0 + ((lane < 16) ? 0 : 8)];
                v8h b0 = *(const v8h*)bp_;
                v8h b1 = *(const v8h*)(bp_ + 16);
                v16h bfr;
#pragma unroll
                for (int q = 0; q < 8; ++q) { bfr[q] = b0[q]; bfr[q + 8] = b1[q]; }
                acc[i] = __builtin_amdgcn_wmma_f32_16x16x32_f16(
                    false, a, false, bfr, (short)0, acc[i], false, false);
            }
        }
#pragma unroll
        for (int i = 0; i < 2; ++i) {
            int N = wave * 32 + i * 16 + (lane & 15);
            float bpv = bp[N];
#pragma unroll
            for (int j = 0; j < 8; ++j) {
                int M = j + ((lane < 16) ? 0 : 8);
                y[((size_t)(t * RPT + rbase + M)) * C_ + N] = acc[i][j] + bpv;
            }
        }
    }
}

extern "C" void kernel_launch(void* const* d_in, const int* in_sizes, int n_in,
                              void* d_out, int out_size, void* d_ws, size_t ws_size,
                              hipStream_t stream) {
    const float* x   = (const float*)d_in[0];
    const float* Wq  = (const float*)d_in[1];
    const float* Wk  = (const float*)d_in[2];
    const float* Wp  = (const float*)d_in[3];
    const float* bp  = (const float*)d_in[4];
    const float* pos = (const float*)d_in[5];

    _Float16* ws     = (_Float16*)d_ws;
    _Float16* Wq_h   = ws + WQ_OFF;
    _Float16* Wk_h   = ws + WK_OFF;
    _Float16* Wp_h   = ws + WP_OFF;
    _Float16* pos_h  = ws + POS_OFF;
    _Float16* attn_h = ws + ATTN_OFF;

    float* y    = (float*)d_out;
    float* aout = y + (size_t)T_ * RPT * C_;   // second tuple element

    cvt_kernel <<<1024, 256, 0, stream>>>(Wq, Wk, Wp, pos, ws);
    qk_kernel  <<<RPT / NROW, 256, 0, stream>>>(x, Wq_h, Wk_h, pos_h, attn_h);
    proj_kernel<<<RPT / NROW, 256, 0, stream>>>(attn_h, Wp_h, bp, y, aout);
}
